// Model_45904610459674
// MI455X (gfx1250) — matrix operand order repaired and checked by reading, not compiled, not run on hardware
//
#include <hip/hip_runtime.h>
#include <math.h>

typedef __attribute__((ext_vector_type(16))) _Float16 v16h;
typedef __attribute__((ext_vector_type(8)))  float    v8f;

// ---------------- model constants ----------------
#define BATCH 128
#define SEQL  96
#define DECL  144
#define DM    512
#define NH    8
#define EH    64
#define DFF   2048
#define CCH   7
#define NFM   4
#define MENC  48
#define MDEC  64
#define MKV   48

// =======================================================================
// WMMA GEMM:  C[M,N] = A[M,K] @ W[N,K]^T (+bias) (+GELU) (+residual)
// One wave per 32x32 tile (2x2 register blocking of 16x16 WMMA tiles),
// K-step 32, f32->f16 on the fly, f32 accumulate. Each A/B fragment is
// reused by two WMMAs.
// =======================================================================

// A fragment (16x32 f16): p already points at row base + 8*hsel.
// elements 0..7 <- p[0..7], elements 8..15 <- p[16..23]
__device__ __forceinline__ v16h load_a_frag(const float* __restrict__ p)
{
    const float4* q0 = reinterpret_cast<const float4*>(p);
    const float4* q1 = reinterpret_cast<const float4*>(p + 16);
    float4 a0 = q0[0], a1 = q0[1], a2 = q1[0], a3 = q1[1];
    v16h v;
    v[0]=(_Float16)a0.x;  v[1]=(_Float16)a0.y;  v[2]=(_Float16)a0.z;  v[3]=(_Float16)a0.w;
    v[4]=(_Float16)a1.x;  v[5]=(_Float16)a1.y;  v[6]=(_Float16)a1.z;  v[7]=(_Float16)a1.w;
    v[8]=(_Float16)a2.x;  v[9]=(_Float16)a2.y;  v[10]=(_Float16)a2.z; v[11]=(_Float16)a2.w;
    v[12]=(_Float16)a3.x; v[13]=(_Float16)a3.y; v[14]=(_Float16)a3.z; v[15]=(_Float16)a3.w;
    return v;
}

// B fragment (32x16 f16): p already points at W row base + 16*hsel.
// elements 0..15 <- p[0..15]
__device__ __forceinline__ v16h load_b_frag(const float* __restrict__ p)
{
    const float4* q = reinterpret_cast<const float4*>(p);
    float4 b0 = q[0], b1 = q[1], b2 = q[2], b3 = q[3];
    v16h v;
    v[0]=(_Float16)b0.x;  v[1]=(_Float16)b0.y;  v[2]=(_Float16)b0.z;  v[3]=(_Float16)b0.w;
    v[4]=(_Float16)b1.x;  v[5]=(_Float16)b1.y;  v[6]=(_Float16)b1.z;  v[7]=(_Float16)b1.w;
    v[8]=(_Float16)b2.x;  v[9]=(_Float16)b2.y;  v[10]=(_Float16)b2.z; v[11]=(_Float16)b2.w;
    v[12]=(_Float16)b3.x; v[13]=(_Float16)b3.y; v[14]=(_Float16)b3.z; v[15]=(_Float16)b3.w;
    return v;
}

__global__ void __launch_bounds__(32)
wmma_gemm_kernel(const float* __restrict__ A, const float* __restrict__ W,
                 const float* __restrict__ bias, const float* __restrict__ resid,
                 float* __restrict__ C, int M, int N, int K, int act)
{
    const int lane  = threadIdx.x;
    const int hsel  = lane >> 4;     // 0 or 1
    const int lmod  = lane & 15;
    const int tileN = blockIdx.x << 5;   // 32-wide
    const int tileM = blockIdx.y << 5;   // 32-tall

    // A fragment rows: lanes 0-15 get K offsets {0..7,16..23},
    // lanes 16-31 get {8..15,24..31}  (ISA 16-bit A 16x32 layout)
    const float* arow0 = A + (size_t)(tileM + lmod) * K + 8 * hsel;
    const float* arow1 = arow0 + (size_t)16 * K;
    // B fragment cols: lanes 0-15 K {0..7,16..23}, lanes 16-31 K {8..15,24..31}
    const float* brow0 = W + (size_t)(tileN + lmod) * K + 8 * hsel;
    const float* brow1 = brow0 + (size_t)16 * K;

    v8f acc00 = {}, acc01 = {}, acc10 = {}, acc11 = {};
    for (int k0 = 0; k0 < K; k0 += 32) {
        if (k0 + 32 < K) {   // prefetch next K-tile of A (global_prefetch_b8)
            __builtin_prefetch(arow0 + k0 + 32, 0, 1);
            __builtin_prefetch(arow1 + k0 + 32, 0, 1);
        }
        v16h a0 = load_a_frag(arow0 + k0);
        v16h a1 = load_a_frag(arow1 + k0);
        v16h b0 = load_a_frag(brow0 + k0);
        v16h b1 = load_a_frag(brow1 + k0);
        acc00 = __builtin_amdgcn_wmma_f32_16x16x32_f16(false, a0, false, b0, (short)0, acc00, false, false);
        acc01 = __builtin_amdgcn_wmma_f32_16x16x32_f16(false, a0, false, b1, (short)0, acc01, false, false);
        acc10 = __builtin_amdgcn_wmma_f32_16x16x32_f16(false, a1, false, b0, (short)0, acc10, false, false);
        acc11 = __builtin_amdgcn_wmma_f32_16x16x32_f16(false, a1, false, b1, (short)0, acc11, false, false);
    }

    // epilogue: C/D layout -> element r of acc: row = 16*i + r + 8*hsel,
    // col = 16*j + lmod within the 32x32 tile
    const v8f* accs[4] = { &acc00, &acc01, &acc10, &acc11 };
#pragma unroll
    for (int i = 0; i < 2; ++i) {
#pragma unroll
        for (int j = 0; j < 2; ++j) {
            const v8f& a = *accs[i * 2 + j];
            const int col = tileN + 16 * j + lmod;
            const float bval = bias ? bias[col] : 0.0f;
#pragma unroll
            for (int r = 0; r < 8; ++r) {
                const int row = tileM + 16 * i + r + 8 * hsel;
                float v = a[r] + bval;
                if (act) v = 0.5f * v * (1.0f + erff(v * 0.70710678118654752f));
                if (resid) v += resid[(size_t)row * N + col];
                C[(size_t)row * N + col] = v;
            }
        }
    }
}

// =======================================================================
// series_decomp init on x_enc -> seasonal_init [B,144,7], trend_init [B,144,7]
// =======================================================================
__global__ void decomp_init_kernel(const float* __restrict__ x, float* __restrict__ sea,
                                   float* __restrict__ tre, int total)
{
    int idx = blockIdx.x * blockDim.x + threadIdx.x;
    if (idx >= total) return;
    int c = idx % CCH, l = (idx / CCH) % DECL, b = idx / (CCH * DECL);
    const float* xb = x + (size_t)b * SEQL * CCH + c;
    if (l < 48) {
        int le = l + 48;
        float s = 0.f;
        for (int j = -12; j <= 12; ++j) {
            int ll = le + j; ll = ll < 0 ? 0 : (ll > SEQL - 1 ? SEQL - 1 : ll);
            s += xb[ll * CCH];
        }
        float ma = s * (1.0f / 25.0f);
        sea[idx] = xb[le * CCH] - ma;
        tre[idx] = ma;
    } else {
        float m = 0.f;
        for (int ll = 0; ll < SEQL; ++ll) m += xb[ll * CCH];
        sea[idx] = 0.f;
        tre[idx] = m * (1.0f / (float)SEQL);
    }
}

// =======================================================================
// embed: circular conv (k=3) over C=7 + time-feature linear
// =======================================================================
__global__ void embed_kernel(const float* __restrict__ x, const float* __restrict__ mark,
                             const float* __restrict__ cw, const float* __restrict__ tw,
                             float* __restrict__ out, int L, int total)
{
    int idx = blockIdx.x * blockDim.x + threadIdx.x;
    if (idx >= total) return;
    int d = idx % DM, l = (idx / DM) % L, b = idx / (DM * L);
    int lm = (l == 0) ? L - 1 : l - 1;
    int lp = (l == L - 1) ? 0 : l + 1;
    const float* xb = x + (size_t)b * L * CCH;
    const float* w  = cw + d * CCH * 3;
    float s = 0.f;
#pragma unroll
    for (int c = 0; c < CCH; ++c)
        s += xb[lm*CCH+c]*w[c*3+0] + xb[l*CCH+c]*w[c*3+1] + xb[lp*CCH+c]*w[c*3+2];
    const float* mb = mark + ((size_t)b * L + l) * NFM;
    const float* twd = tw + d * NFM;
#pragma unroll
    for (int f = 0; f < NFM; ++f) s += mb[f] * twd[f];
    out[idx] = s;
}

// =======================================================================
// truncated rfft (first M bins) of [B,L,512] along L (per d) -> complex
// incremental twiddle rotation, 1 sincos per thread
// =======================================================================
__global__ void dft_kernel(const float* __restrict__ src, float* __restrict__ outc,
                           int L, int M, int total)
{
    int idx = blockIdx.x * blockDim.x + threadIdx.x;
    if (idx >= total) return;
    int m = idx % M, oe = (idx / M) % DM, b = idx / (M * DM);
    const float* p = src + (size_t)b * L * DM + oe;
    float ang = -6.283185307179586f * (float)m / (float)L;
    float wi, wr; sincosf(ang, &wi, &wr);
    float pr = 1.f, pi = 0.f, sr = 0.f, si = 0.f;
    for (int l = 0; l < L; ++l) {
        float v = p[(size_t)l * DM];
        sr = fmaf(v, pr, sr); si = fmaf(v, pi, si);
        float npr = pr * wr - pi * wi;
        pi = fmaf(pr, wi, pi * wr);
        pr = npr;
    }
    outc[2*idx] = sr; outc[2*idx+1] = si;
}

// =======================================================================
// mode mixing: sel[b,h,o,m] = sum_e xft[b,h,e,m] * w[h,e,o,m]  (complex)
// =======================================================================
__global__ void modemix_kernel(const float* __restrict__ xft, const float* __restrict__ w,
                               float* __restrict__ sel, int M, int total)
{
    int idx = blockIdx.x * blockDim.x + threadIdx.x;
    if (idx >= total) return;
    int m = idx % M, o = (idx / M) % EH, h = (idx / (M * EH)) % NH, b = idx / (M * EH * NH);
    const float* xp = xft + 2 * ((size_t)(b * NH + h) * EH * M + m);
    const float* wp = w   + 2 * ((size_t)h * EH * EH * M + (size_t)o * M + m);
    const int xs = 2 * M, wsd = 2 * EH * M;
    float sr = 0.f, si = 0.f;
    for (int e = 0; e < EH; ++e) {
        float ar = xp[e*xs], ai = xp[e*xs+1];
        float br = wp[e*wsd], bi = wp[e*wsd+1];
        sr += ar*br - ai*bi;
        si += ar*bi + ai*br;
    }
    sel[2*idx] = sr; sel[2*idx+1] = si;
}

// =======================================================================
// truncated irfft: bins 0..M-1 (M < L/2), DC imag ignored -> real [B,512,L]
// =======================================================================
__global__ void irfft_kernel(const float* __restrict__ sel, float* __restrict__ out,
                             int L, int M, float scale, int total)
{
    int idx = blockIdx.x * blockDim.x + threadIdx.x;
    if (idx >= total) return;
    int l = idx % L, oe = (idx / L) % DM, b = idx / (L * DM);
    const float* s = sel + 2 * ((size_t)(b * DM + oe) * M);
    float ang = 6.283185307179586f * (float)l / (float)L;
    float wi, wr; sincosf(ang, &wi, &wr);
    float pr = wr, pi = wi;
    float acc = s[0];                       // Re(X0); imag of DC ignored (C2R)
    for (int m = 1; m < M; ++m) {
        acc = fmaf( 2.f * s[2*m],   pr, acc);
        acc = fmaf(-2.f * s[2*m+1], pi, acc);
        float npr = pr * wr - pi * wi;
        pi = fmaf(pr, wi, pi * wr);
        pr = npr;
    }
    out[idx] = acc * scale / (float)L;
}

// =======================================================================
// fourier_cross pieces
// =======================================================================
__global__ void xqk_kernel(const float* __restrict__ qft, const float* __restrict__ kft,
                           float* __restrict__ xqk, int total)
{
    int idx = blockIdx.x * blockDim.x + threadIdx.x;
    if (idx >= total) return;                      // total = B*H*64*48
    int y = idx % MKV, x = (idx / MKV) % MDEC, bh = idx / (MKV * MDEC);
    const float* qp = qft + (size_t)bh * EH * MDEC * 2 + x * 2;  // e-stride 2*MDEC
    const float* kp = kft + (size_t)bh * EH * MKV  * 2 + y * 2;  // e-stride 2*MKV
    float sr = 0.f, si = 0.f;
    for (int e = 0; e < EH; ++e) {
        float ar = qp[e*2*MDEC], ai = qp[e*2*MDEC+1];
        float br = kp[e*2*MKV],  bi = kp[e*2*MKV+1];
        sr += ar*br - ai*bi;
        si += ar*bi + ai*br;
    }
    // complex tanh
    float tr, ti;
    if (fabsf(sr) > 15.f) { tr = copysignf(1.f, sr); ti = 0.f; }
    else {
        float den = coshf(2.f*sr) + cosf(2.f*si);
        tr = sinhf(2.f*sr) / den;
        ti = sinf(2.f*si) / den;
    }
    xqk[2*idx] = tr; xqk[2*idx+1] = ti;
}

__global__ void xqkv_kernel(const float* __restrict__ xqk, const float* __restrict__ kft,
                            float* __restrict__ xqkv, int total)
{
    int idx = blockIdx.x * blockDim.x + threadIdx.x;
    if (idx >= total) return;                      // total = B*H*E*64
    int x = idx % MDEC, e = (idx / MDEC) % EH, bh = idx / (MDEC * EH);
    const float* qk = xqk + ((size_t)bh * MDEC + x) * MKV * 2;
    const float* kp = kft + ((size_t)bh * EH   + e) * MKV * 2;
    float sr = 0.f, si = 0.f;
    for (int y = 0; y < MKV; ++y) {
        float ar = qk[2*y], ai = qk[2*y+1];
        float br = kp[2*y], bi = kp[2*y+1];
        sr += ar*br - ai*bi;
        si += ar*bi + ai*br;
    }
    xqkv[2*idx] = sr; xqkv[2*idx+1] = si;
}

__global__ void xqkvw_kernel(const float* __restrict__ xqkv, const float* __restrict__ w,
                             float* __restrict__ sel, int total)
{
    int idx = blockIdx.x * blockDim.x + threadIdx.x;
    if (idx >= total) return;                      // total = B*H*64(o)*64(x)
    int x = idx % MDEC, o = (idx / MDEC) % EH, h = (idx / (MDEC * EH)) % NH,
        b = idx / (MDEC * EH * NH);
    const float* vp = xqkv + ((size_t)(b * NH + h) * EH) * MDEC * 2 + x * 2; // e-str 2*64
    const float* wp = w + 2 * ((size_t)h * EH * EH * MDEC + (size_t)o * MDEC + x);
    const int vs = 2 * MDEC, wsd = 2 * EH * MDEC;
    float sr = 0.f, si = 0.f;
    for (int e = 0; e < EH; ++e) {
        float ar = vp[e*vs], ai = vp[e*vs+1];
        float br = wp[e*wsd], bi = wp[e*wsd+1];
        sr += ar*br - ai*bi;
        si += ar*bi + ai*br;
    }
    sel[2*idx] = sr; sel[2*idx+1] = si;
}

// =======================================================================
// series_decomp on [B,L,512]; mode 0: seasonal only, 1: write trend, 2: += trend
// =======================================================================
__global__ void decomp_kernel(const float* __restrict__ x, float* __restrict__ sea,
                              float* __restrict__ tre, int L, int mode, int total)
{
    int idx = blockIdx.x * blockDim.x + threadIdx.x;
    if (idx >= total) return;
    int d = idx % DM, l = (idx / DM) % L, b = idx / (DM * L);
    const float* xb = x + (size_t)b * L * DM + d;
    float s = 0.f;
#pragma unroll
    for (int j = -12; j <= 12; ++j) {
        int ll = l + j; ll = ll < 0 ? 0 : (ll >= L ? L - 1 : ll);
        s += xb[(size_t)ll * DM];
    }
    float mean = s * (1.0f / 25.0f);
    sea[idx] = x[idx] - mean;
    if (mode == 1) tre[idx] = mean;
    else if (mode == 2) tre[idx] += mean;
}

// =======================================================================
// my_layernorm: phase A per (b,l), phase B subtract per-(b,d) time mean
// =======================================================================
__global__ void ln_row_kernel(const float* __restrict__ x, const float* __restrict__ w,
                              const float* __restrict__ bia, float* __restrict__ out, int rows)
{
    int r = blockIdx.x * blockDim.x + threadIdx.x;
    if (r >= rows) return;
    const float* xr = x + (size_t)r * DM;
    float mu = 0.f;
    for (int d = 0; d < DM; ++d) mu += xr[d];
    mu *= (1.0f / (float)DM);
    float var = 0.f;
    for (int d = 0; d < DM; ++d) { float t = xr[d] - mu; var += t * t; }
    var *= (1.0f / (float)DM);
    float inv = rsqrtf(var + 1e-5f);
    float* orow = out + (size_t)r * DM;
    for (int d = 0; d < DM; ++d) orow[d] = (xr[d] - mu) * inv * w[d] + bia[d];
}

__global__ void ln_time_kernel(const float* __restrict__ xh, float* __restrict__ out,
                               int L, int total)
{
    int idx = blockIdx.x * blockDim.x + threadIdx.x;
    if (idx >= total) return;                      // total = B*512
    int d = idx % DM, b = idx / DM;
    const float* p = xh + (size_t)b * L * DM + d;
    float m = 0.f;
    for (int l = 0; l < L; ++l) m += p[(size_t)l * DM];
    m /= (float)L;
    float* o = out + (size_t)b * L * DM + d;
    for (int l = 0; l < L; ++l) o[(size_t)l * DM] = p[(size_t)l * DM] - m;
}

// =======================================================================
// final: trend_init + circ_conv(t1+t2+t3, trend_w) + xln @ proj_w.T + proj_b
// only last PRED=96 positions
// =======================================================================
__global__ void final_kernel(const float* __restrict__ xln, const float* __restrict__ tsum,
                             const float* __restrict__ tinit, const float* __restrict__ tw,
                             const float* __restrict__ pw, const float* __restrict__ pb,
                             float* __restrict__ out, int total)
{
    int idx = blockIdx.x * blockDim.x + threadIdx.x;
    if (idx >= total) return;                      // total = 128*96*7
    int c = idx % CCH, lo = (idx / CCH) % 96, b = idx / (CCH * 96);
    int l = lo + 48;
    int lm = l - 1;
    int lp = (l == DECL - 1) ? 0 : l + 1;
    const float* t = tsum + (size_t)b * DECL * DM;
    const float* w = tw + c * DM * 3;
    float conv = 0.f;
    for (int d = 0; d < DM; ++d) {
        conv = fmaf(t[(size_t)lm*DM+d], w[d*3+0], conv);
        conv = fmaf(t[(size_t)l *DM+d], w[d*3+1], conv);
        conv = fmaf(t[(size_t)lp*DM+d], w[d*3+2], conv);
    }
    float se = pb[c];
    const float* xl = xln + ((size_t)b * DECL + l) * DM;
    const float* pwc = pw + c * DM;
    for (int d = 0; d < DM; ++d) se = fmaf(xl[d], pwc[d], se);
    out[idx] = tinit[((size_t)b * DECL + l) * CCH + c] + conv + se;
}

// =======================================================================
// host side
// =======================================================================
static inline void gemm(const float* A, const float* W, const float* bias,
                        const float* resid, float* C, int M, int N, int K,
                        int act, hipStream_t s)
{
    dim3 g(N / 32, M / 32);
    wmma_gemm_kernel<<<g, dim3(32), 0, s>>>(A, W, bias, resid, C, M, N, K, act);
}
static inline int nblk(int t) { return (t + 255) / 256; }

extern "C" void kernel_launch(void* const* d_in, const int* in_sizes, int n_in,
                              void* d_out, int out_size, void* d_ws, size_t ws_size,
                              hipStream_t stream)
{
    (void)in_sizes; (void)n_in; (void)out_size; (void)ws_size;
    const float* x_enc      = (const float*)d_in[0];
    const float* mark_enc   = (const float*)d_in[1];
    const float* mark_dec   = (const float*)d_in[3];
    const float* enc_conv_w = (const float*)d_in[4];
    const float* enc_time_w = (const float*)d_in[5];
    const float* dec_conv_w = (const float*)d_in[6];
    const float* dec_time_w = (const float*)d_in[7];
    const float* enc_qw = (const float*)d_in[8];
    const float* enc_qb = (const float*)d_in[9];
    // enc_kw/kb (10,11), enc_vw/vb (12,13): dead in fourier_block -> skipped
    const float* enc_ow = (const float*)d_in[14];
    const float* enc_ob = (const float*)d_in[15];
    const float* enc_fw = (const float*)d_in[16];
    const float* enc_c1 = (const float*)d_in[17];
    const float* enc_c2 = (const float*)d_in[18];
    const float* enc_nw = (const float*)d_in[19];
    const float* enc_nb = (const float*)d_in[20];
    const float* ds_qw  = (const float*)d_in[21];
    const float* ds_qb  = (const float*)d_in[22];
    // ds_kw/kb/vw/vb (23-26): dead -> skipped
    const float* ds_ow  = (const float*)d_in[27];
    const float* ds_ob  = (const float*)d_in[28];
    const float* ds_fw  = (const float*)d_in[29];
    const float* dc_qw  = (const float*)d_in[30];
    const float* dc_qb  = (const float*)d_in[31];
    const float* dc_kw  = (const float*)d_in[32];
    const float* dc_kb  = (const float*)d_in[33];
    // dc_vw/vb (34,35): dead -> skipped
    const float* dc_ow  = (const float*)d_in[36];
    const float* dc_ob  = (const float*)d_in[37];
    const float* dc_fw  = (const float*)d_in[38];
    const float* dec_c1 = (const float*)d_in[39];
    const float* dec_c2 = (const float*)d_in[40];
    const float* trend_w= (const float*)d_in[41];
    const float* dec_nw = (const float*)d_in[42];
    const float* dec_nb = (const float*)d_in[43];
    const float* proj_w = (const float*)d_in[44];
    const float* proj_b = (const float*)d_in[45];
    float* out = (float*)d_out;

    // ---- workspace arena (floats) ----
    float* ws = (float*)d_ws;
    float* S0 = ws;                       // 18432*2048 = 37,748,736 (FFN / cross temps)
    float* S1 = S0 + 37748736;            // 18432*512
    float* S2 = S1 + 9437184;
    float* S3 = S2 + 9437184;
    float* S4 = S3 + 9437184;             // trend sum t1+t2+t3
    float* S5 = S4 + 9437184;             // enc_out 12288*512
    float* C0 = S5 + 6291456;             // complex scratch 8,388,608
    float* C1 = C0 + 8388608;             // complex scratch 8,388,608
    float* T0 = C1 + 8388608;             // seasonal_init 128*144*7
    float* T1 = T0 + 129024;              // trend_init   128*144*7
    // cross-attn complex temps packed inside S0:
    float* QFT  = S0;                     // 8,388,608
    float* KFT  = S0 + 8388608;           // 6,291,456
    float* XQK  = S0 + 14680064;          // 6,291,456
    float* XQKV = S0 + 20971520;          // 8,388,608

    const int MROWS_E = BATCH * SEQL;     // 12288
    const int MROWS_D = BATCH * DECL;     // 18432

    // ---- init decomposition ----
    decomp_init_kernel<<<nblk(129024), 256, 0, stream>>>(x_enc, T0, T1, 129024);

    // ---- encoder ----
    {
        int tot = BATCH * SEQL * DM;
        embed_kernel<<<nblk(tot), 256, 0, stream>>>(x_enc, mark_enc, enc_conv_w,
                                                    enc_time_w, S1, SEQL, tot);
    }
    for (int l = 0; l < 2; ++l) {
        const float* qw = enc_qw + (size_t)l * DM * DM;
        const float* qb = enc_qb + (size_t)l * DM;
        const float* ow = enc_ow + (size_t)l * DM * DM;
        const float* ob = enc_ob + (size_t)l * DM;
        const float* fw = enc_fw + (size_t)l * NH * EH * EH * MENC * 2;
        const float* c1 = enc_c1 + (size_t)l * DFF * DM;
        const float* c2 = enc_c2 + (size_t)l * DM * DFF;

        gemm(S1, qw, qb, nullptr, S2, MROWS_E, DM, DM, 0, stream);
        { int t = BATCH * DM * MENC;
          dft_kernel<<<nblk(t), 256, 0, stream>>>(S2, C0, SEQL, MENC, t); }
        { int t = BATCH * NH * EH * MENC;
          modemix_kernel<<<nblk(t), 256, 0, stream>>>(C0, fw, C1, MENC, t); }
        { int t = BATCH * DM * SEQL;
          irfft_kernel<<<nblk(t), 256, 0, stream>>>(C1, S3, SEQL, MENC, 1.0f, t); }
        gemm(S3, ow, ob, S1, S2, MROWS_E, DM, DM, 0, stream);   // + residual x
        { int t = BATCH * SEQL * DM;
          decomp_kernel<<<nblk(t), 256, 0, stream>>>(S2, S1, nullptr, SEQL, 0, t); }
        gemm(S1, c1, nullptr, nullptr, S0, MROWS_E, DFF, DM, 1, stream);  // GELU
        gemm(S0, c2, nullptr, S1, S2, MROWS_E, DM, DFF, 0, stream);       // + residual
        { int t = BATCH * SEQL * DM;
          decomp_kernel<<<nblk(t), 256, 0, stream>>>(S2, S1, nullptr, SEQL, 0, t); }
    }
    ln_row_kernel<<<nblk(MROWS_E), 256, 0, stream>>>(S1, enc_nw, enc_nb, S2, MROWS_E);
    { int t = BATCH * DM;
      ln_time_kernel<<<nblk(t), 256, 0, stream>>>(S2, S5, SEQL, t); }   // enc_out

    // ---- decoder ----
    { int tot = BATCH * DECL * DM;
      embed_kernel<<<nblk(tot), 256, 0, stream>>>(T0, mark_dec, dec_conv_w,
                                                  dec_time_w, S1, DECL, tot); }
    // self fourier attention
    gemm(S1, ds_qw, ds_qb, nullptr, S2, MROWS_D, DM, DM, 0, stream);
    { int t = BATCH * DM * MDEC;
      dft_kernel<<<nblk(t), 256, 0, stream>>>(S2, C0, DECL, MDEC, t); }
    { int t = BATCH * NH * EH * MDEC;
      modemix_kernel<<<nblk(t), 256, 0, stream>>>(C0, ds_fw, C1, MDEC, t); }
    { int t = BATCH * DM * DECL;
      irfft_kernel<<<nblk(t), 256, 0, stream>>>(C1, S3, DECL, MDEC, 1.0f, t); }
    gemm(S3, ds_ow, ds_ob, S1, S2, MROWS_D, DM, DM, 0, stream);
    { int t = BATCH * DECL * DM;
      decomp_kernel<<<nblk(t), 256, 0, stream>>>(S2, S1, S4, DECL, 1, t); }  // t1 (write)

    // cross fourier attention
    gemm(S5, dc_kw, dc_kb, nullptr, S3, MROWS_E, DM, DM, 0, stream);   // k = enc_out proj
    { int t = BATCH * DM * MKV;
      dft_kernel<<<nblk(t), 256, 0, stream>>>(S3, KFT, SEQL, MKV, t); }
    gemm(S1, dc_qw, dc_qb, nullptr, S2, MROWS_D, DM, DM, 0, stream);
    { int t = BATCH * DM * MDEC;
      dft_kernel<<<nblk(t), 256, 0, stream>>>(S2, QFT, DECL, MDEC, t); }
    { int t = BATCH * NH * MDEC * MKV;
      xqk_kernel<<<nblk(t), 256, 0, stream>>>(QFT, KFT, XQK, t); }
    { int t = BATCH * NH * EH * MDEC;
      xqkv_kernel<<<nblk(t), 256, 0, stream>>>(XQK, KFT, XQKV, t); }
    { int t = BATCH * NH * EH * MDEC;
      xqkvw_kernel<<<nblk(t), 256, 0, stream>>>(XQKV, dc_fw, C0, t); }
    { int t = BATCH * DM * DECL;
      irfft_kernel<<<nblk(t), 256, 0, stream>>>(C0, S3, DECL, MDEC,
                                                1.0f / ((float)DM * (float)DM), t); }
    gemm(S3, dc_ow, dc_ob, S1, S2, MROWS_D, DM, DM, 0, stream);
    { int t = BATCH * DECL * DM;
      decomp_kernel<<<nblk(t), 256, 0, stream>>>(S2, S1, S4, DECL, 2, t); } // t2 (acc)

    // FFN
    gemm(S1, dec_c1, nullptr, nullptr, S0, MROWS_D, DFF, DM, 1, stream);
    gemm(S0, dec_c2, nullptr, S1, S2, MROWS_D, DM, DFF, 0, stream);
    { int t = BATCH * DECL * DM;
      decomp_kernel<<<nblk(t), 256, 0, stream>>>(S2, S1, S4, DECL, 2, t); } // t3 (acc)

    // layernorm + final
    ln_row_kernel<<<nblk(MROWS_D), 256, 0, stream>>>(S1, dec_nw, dec_nb, S2, MROWS_D);
    { int t = BATCH * DM;
      ln_time_kernel<<<nblk(t), 256, 0, stream>>>(S2, S3, DECL, t); }
    { int t = BATCH * 96 * CCH;
      final_kernel<<<nblk(t), 256, 0, stream>>>(S3, S4, T1, trend_w, proj_w, proj_b,
                                                out, t); }
}
